// SwitchFFN_58222576665158
// MI455X (gfx1250) — matrix operand order repaired and checked by reading, not compiled, not run on hardware
//
#include <hip/hip_runtime.h>

typedef _Float16 v16h __attribute__((ext_vector_type(16)));
typedef _Float16 v8h  __attribute__((ext_vector_type(8)));
typedef _Float16 v4h  __attribute__((ext_vector_type(4)));
typedef float    v8f  __attribute__((ext_vector_type(8)));
typedef float    f32x4 __attribute__((ext_vector_type(4)));
typedef unsigned int u32x4 __attribute__((ext_vector_type(4)));
typedef int      i32x4 __attribute__((ext_vector_type(4)));
typedef int      i32x8 __attribute__((ext_vector_type(8)));

static constexpr int T_TOK = 8192;
static constexpr int D     = 1024;
static constexpr int DFF   = 4096;
static constexpr int E     = 8;
static constexpr int CAP   = 1024;   // ceil(T/E * 1.0)

#if defined(__gfx1250__) && __has_builtin(__builtin_amdgcn_tensor_load_to_lds) && \
    __has_builtin(__builtin_amdgcn_s_wait_tensorcnt)
#define MOE_USE_TDM 1
#else
#define MOE_USE_TDM 0
#endif

// ---------------------------------------------------------------------------
// 0) zero the dispatch buffer (unfilled capacity slots must be benign)
// ---------------------------------------------------------------------------
__global__ void moe_zero_kernel(unsigned int* __restrict__ w, size_t nwords) {
    size_t i = (size_t)blockIdx.x * blockDim.x + threadIdx.x;
    size_t stride = (size_t)gridDim.x * blockDim.x;
    for (; i < nwords; i += stride) w[i] = 0u;
}

// ---------------------------------------------------------------------------
// 0b) one-time weight convert+transpose: W [E][K][N] f32 -> Wt [E][N][K] f16
//     (LDS-tiled 32x32 transpose; coalesced read and write)
// ---------------------------------------------------------------------------
__global__ __launch_bounds__(256) void moe_wt_kernel(
    const float* __restrict__ W, _Float16* __restrict__ Wt, int K, int N)
{
    __shared__ _Float16 tile[32][33];
    const int e  = blockIdx.z;
    const int k0 = blockIdx.y * 32;
    const int n0 = blockIdx.x * 32;
    const int tx = threadIdx.x & 31;
    const int ty = threadIdx.x >> 5;            // 8 rows per pass
#pragma unroll
    for (int r = ty; r < 32; r += 8)
        tile[r][tx] = (_Float16)W[((size_t)e * K + k0 + r) * N + n0 + tx];
    __syncthreads();
#pragma unroll
    for (int r = ty; r < 32; r += 8)
        Wt[((size_t)e * N + n0 + r) * K + k0 + tx] = tile[tx][r];
}

// ---------------------------------------------------------------------------
// 1) router: one block per token. logits = x_t . Wr_e, softmax, top1, lse^2
// ---------------------------------------------------------------------------
__global__ __launch_bounds__(256) void moe_router_kernel(
    const float* __restrict__ x, const float* __restrict__ Wr,
    float* __restrict__ top1_prob, int* __restrict__ top1_idx,
    float* __restrict__ probs8, float* __restrict__ lse2_tok)
{
    const int t = blockIdx.x;
    const int tid = threadIdx.x;
    const float* xt = x + (size_t)t * D;

    float xv[4];
#pragma unroll
    for (int i = 0; i < 4; ++i) xv[i] = xt[tid + i * 256];

    float part[E];
#pragma unroll
    for (int e = 0; e < E; ++e) {
        float s = 0.f;
#pragma unroll
        for (int i = 0; i < 4; ++i) s += xv[i] * Wr[e * D + tid + i * 256];
        part[e] = s;
    }

    __shared__ float red[256];
    __shared__ float logits[E];
    for (int e = 0; e < E; ++e) {
        red[tid] = part[e];
        __syncthreads();
        for (int s = 128; s > 0; s >>= 1) {
            if (tid < s) red[tid] += red[tid + s];
            __syncthreads();
        }
        if (tid == 0) logits[e] = red[0];
        __syncthreads();
    }

    if (tid == 0) {
        float mx = logits[0]; int mi = 0;
#pragma unroll
        for (int e = 1; e < E; ++e) if (logits[e] > mx) { mx = logits[e]; mi = e; }
        float pr[E]; float se = 0.f;
#pragma unroll
        for (int e = 0; e < E; ++e) { pr[e] = __expf(logits[e] - mx); se += pr[e]; }
        float inv = 1.f / se;
#pragma unroll
        for (int e = 0; e < E; ++e) { pr[e] *= inv; probs8[(size_t)t * E + e] = pr[e]; }
        float lse = mx + __logf(se);
        lse2_tok[t] = lse * lse;
        top1_prob[t] = pr[mi];
        top1_idx[t]  = mi;
    }
}

// ---------------------------------------------------------------------------
// 2) deterministic stats reduction: Pi sums and sum(lse^2) (fixed-order tree)
// ---------------------------------------------------------------------------
__global__ __launch_bounds__(256) void moe_stats_kernel(
    const float* __restrict__ probs8, const float* __restrict__ lse2_tok,
    float* __restrict__ Pi_sum, float* __restrict__ lse2_sum)
{
    const int tid = threadIdx.x;
    float local[E + 1];
#pragma unroll
    for (int q = 0; q < E + 1; ++q) local[q] = 0.f;
    const int per = T_TOK / 256;
    for (int i = 0; i < per; ++i) {
        int t = tid * per + i;
#pragma unroll
        for (int e = 0; e < E; ++e) local[e] += probs8[(size_t)t * E + e];
        local[E] += lse2_tok[t];
    }
    __shared__ float red[256];
    for (int q = 0; q < E + 1; ++q) {
        red[tid] = local[q];
        __syncthreads();
        for (int s = 128; s > 0; s >>= 1) {
            if (tid < s) red[tid] += red[tid + s];
            __syncthreads();
        }
        if (tid == 0) { if (q < E) Pi_sum[q] = red[0]; else lse2_sum[0] = red[0]; }
        __syncthreads();
    }
}

// ---------------------------------------------------------------------------
// 3) serial first-come-first-served scan per expert (token order)
// ---------------------------------------------------------------------------
__global__ void moe_scan_kernel(const int* __restrict__ idx, int* __restrict__ pos,
                                int* __restrict__ counts)
{
    int e = threadIdx.x;
    if (e >= E) return;
    int c = 0;
    for (int t = 0; t < T_TOK; ++t) {
        if (idx[t] == e) { pos[t] = c; ++c; }
    }
    counts[e] = c;
}

// ---------------------------------------------------------------------------
// 4) dispatch: scatter fitting tokens into f16 buf[E, C, D]
// ---------------------------------------------------------------------------
__global__ __launch_bounds__(256) void moe_dispatch_kernel(
    const float* __restrict__ x, const int* __restrict__ idx,
    const int* __restrict__ pos, _Float16* __restrict__ buf)
{
    const int t = blockIdx.x;
    const int p = pos[t];
    if (p >= CAP) return;
    const int e = idx[t];
    const float* xt = x + (size_t)t * D;
    _Float16* dst = buf + ((size_t)e * CAP + p) * D;
    const int j = threadIdx.x * 4;
    f32x4 v = *(const f32x4*)(xt + j);
    v4h h;
    h[0] = (_Float16)v[0]; h[1] = (_Float16)v[1];
    h[2] = (_Float16)v[2]; h[3] = (_Float16)v[3];
    *(v4h*)(dst + j) = h;
}

// ---------------------------------------------------------------------------
// TDM helper: 2D f16 tile load (tile_dim0 x tile_dim1) into LDS via D#
// descriptor (CDNA5 ISA 08_async_tensor §8.3/8.4). data_size=2B, no padding,
// no multicast, no iterate. Issued per-wave (EXEC ignored), TENSORcnt-tracked.
// ---------------------------------------------------------------------------
#if MOE_USE_TDM
__device__ __forceinline__ void moe_tdm_load_2d(
    const void* gaddr, unsigned lds_off,
    unsigned tensor_d0, unsigned tensor_d1,
    unsigned tile_d0, unsigned tile_d1,
    unsigned long long d0_stride_elems)
{
    const unsigned long long ga = (unsigned long long)gaddr;
    u32x4 g0;
    g0[0] = 1u;                                            // count=1 (valid), user mode
    g0[1] = lds_off;                                       // lds_addr
    g0[2] = (unsigned)(ga & 0xFFFFFFFFu);                  // global_addr[31:0]
    g0[3] = (unsigned)((ga >> 32) & 0x01FFFFFFu)           // global_addr[56:32]
          | (2u << 30);                                    // type=2 ("image")
    i32x8 g1;
    g1[0] = (int)(1u << 16);                               // data_size=1 -> 2 bytes
    g1[1] = (int)((tensor_d0 & 0xFFFFu) << 16);            // barrier_addr=0 | dim0[15:0]
    g1[2] = (int)(((tensor_d0 >> 16) & 0xFFFFu)
          | ((tensor_d1 & 0xFFFFu) << 16));                // dim0[31:16] | dim1[15:0]
    g1[3] = (int)(((tensor_d1 >> 16) & 0xFFFFu)
          | ((tile_d0 & 0xFFFFu) << 16));                  // dim1[31:16] | tile_dim0
    g1[4] = (int)(tile_d1 & 0xFFFFu);                      // tile_dim1 | tile_dim2=0
    g1[5] = (int)(d0_stride_elems & 0xFFFFFFFFull);        // dim0_stride[31:0]
    g1[6] = (int)((d0_stride_elems >> 32) & 0xFFFFull);    // dim0_stride[47:32]
    g1[7] = 0;
    i32x4 z4 = {0, 0, 0, 0};
#if defined(__clang_major__) && (__clang_major__ >= 23)
    i32x8 z8 = {0, 0, 0, 0, 0, 0, 0, 0};
    __builtin_amdgcn_tensor_load_to_lds(g0, g1, z4, z4, z8, 0);
#else
    __builtin_amdgcn_tensor_load_to_lds(g0, g1, z4, z4, 0);
#endif
}
#endif

// ---------------------------------------------------------------------------
// 5/6) tiled WMMA GEMM per expert: Out[e] = act(A[e] (MxK f16) * Bt[e]^T + bias)
//      Bt pre-transposed [E][N][K] f16 -> both tiles are plain 2D copies.
//      128x128 block tile, 8 waves, 8 wmma accumulators per wave.
//      Tiles staged by the Tensor Data Mover, double-buffered.
// ---------------------------------------------------------------------------
template <bool RELU, bool OUT_F16>
__global__ __launch_bounds__(256) void moe_gemm_kernel(
    const _Float16* __restrict__ A,   // [E, M, K]
    const _Float16* __restrict__ Bt,  // [E, N, K]
    const float* __restrict__ bias,   // [E, N]
    void* __restrict__ Out,           // [E, M, N]
    int M, int K, int N)
{
    const int e  = blockIdx.z;
    const int m0 = blockIdx.y * 128;
    const int n0 = blockIdx.x * 128;
    const _Float16* Ae = A  + (size_t)e * M * K + (size_t)m0 * K;
    const _Float16* Be = Bt + (size_t)e * N * K + (size_t)n0 * K;
    const float*    be = bias + (size_t)e * N;

    __shared__ _Float16 sA[2][128 * 32];   // [m][k], stride 32 halves
    __shared__ _Float16 sB[2][128 * 32];   // [n][k], stride 32 halves

    const int tid    = threadIdx.x;
    const int wave   = tid >> 5;
    const int lane   = tid & 31;
    const int laneHi = lane >> 4;
    const int l16    = lane & 15;

    v8f acc[8] = {};
    int cur = 0;

#if MOE_USE_TDM
    const unsigned offA[2] = { (unsigned)(unsigned long long)(const void*)&sA[0][0],
                               (unsigned)(unsigned long long)(const void*)&sA[1][0] };
    const unsigned offB[2] = { (unsigned)(unsigned long long)(const void*)&sB[0][0],
                               (unsigned)(unsigned long long)(const void*)&sB[1][0] };
    if (wave == 0) {
        moe_tdm_load_2d(Ae, offA[0], (unsigned)K, 128u, 32u, 128u, (unsigned long long)K);
        moe_tdm_load_2d(Be, offB[0], (unsigned)K, 128u, 32u, 128u, (unsigned long long)K);
    }
#endif

    for (int k0 = 0; k0 < K; k0 += 32, cur ^= 1) {
#if MOE_USE_TDM
        if (wave == 0) {
            const int k1 = k0 + 32;
            if (k1 < K) {
                // prefetch next pair into the other buffer (safe: all waves
                // passed the barrier that closed their reads of that buffer)
                moe_tdm_load_2d(Ae + k1, offA[cur ^ 1], (unsigned)K, 128u, 32u, 128u,
                                (unsigned long long)K);
                moe_tdm_load_2d(Be + k1, offB[cur ^ 1], (unsigned)K, 128u, 32u, 128u,
                                (unsigned long long)K);
                __builtin_amdgcn_s_wait_tensorcnt(2);  // current pair landed
            } else {
                __builtin_amdgcn_s_wait_tensorcnt(0);
            }
        }
        __syncthreads();
#else
        // synchronous fallback staging (pure f16 copies)
        {
            const int row = tid >> 1;
            const int seg = (tid & 1) * 16;
            *(f32x4*)&sA[cur][row * 32 + seg] =
                *(const f32x4*)(Ae + (size_t)row * K + k0 + seg);
            *(f32x4*)&sA[cur][row * 32 + seg + 8] =
                *(const f32x4*)(Ae + (size_t)row * K + k0 + seg + 8);
            *(f32x4*)&sB[cur][row * 32 + seg] =
                *(const f32x4*)(Be + (size_t)row * K + k0 + seg);
            *(f32x4*)&sB[cur][row * 32 + seg + 8] =
                *(const f32x4*)(Be + (size_t)row * K + k0 + seg + 8);
        }
        __syncthreads();
#endif

        // A fragment (16x32): lane<16 holds K 0-7 & 16-23; lane>=16 K 8-15 & 24-31
        const _Float16* ap = &sA[cur][(wave * 16 + l16) * 32 + laneHi * 8];
        v8h alo = *(const v8h*)(ap);
        v8h ahi = *(const v8h*)(ap + 16);
        v16h afrag = __builtin_shufflevector(alo, ahi,
            0, 1, 2, 3, 4, 5, 6, 7, 8, 9, 10, 11, 12, 13, 14, 15);

        // 8 n-subtiles: B fragment (32x16): lane<16 holds K 0-7 & 16-23; lane>=16 K 8-15 & 24-31
#pragma unroll
        for (int ns = 0; ns < 8; ++ns) {
            const _Float16* bp = &sB[cur][(ns * 16 + l16) * 32 + laneHi * 8];
            v8h blo = *(const v8h*)(bp);
            v8h bhi = *(const v8h*)(bp + 16);
            v16h bfrag = __builtin_shufflevector(blo, bhi,
                0, 1, 2, 3, 4, 5, 6, 7, 8, 9, 10, 11, 12, 13, 14, 15);
            acc[ns] = __builtin_amdgcn_wmma_f32_16x16x32_f16(
                false, afrag, false, bfrag, (short)0, acc[ns], false, false);
        }
        __syncthreads();
    }

    // epilogue: C/D layout -> row m = r + 8*laneHi, col n = l16
#pragma unroll
    for (int ns = 0; ns < 8; ++ns) {
        const int n  = n0 + ns * 16 + l16;
        const float bv = be[n];
#pragma unroll
        for (int r = 0; r < 8; ++r) {
            const int m = m0 + wave * 16 + laneHi * 8 + r;
            float v = acc[ns][r] + bv;
            if (RELU) v = v > 0.f ? v : 0.f;
            if (OUT_F16)
                ((_Float16*)Out)[(size_t)e * M * N + (size_t)m * N + n] = (_Float16)v;
            else
                ((float*)Out)[(size_t)e * M * N + (size_t)m * N + n] = v;
        }
    }
}

// ---------------------------------------------------------------------------
// 7) combine: routed tokens get prob * expert_out, dropped tokens passthrough
// ---------------------------------------------------------------------------
__global__ __launch_bounds__(256) void moe_combine_kernel(
    const float* __restrict__ x, const int* __restrict__ idx,
    const int* __restrict__ pos, const float* __restrict__ prob,
    const float* __restrict__ eo, float* __restrict__ y)
{
    const int t = blockIdx.x;
    const int p = pos[t];
    const int j = threadIdx.x * 4;
    float* yt = y + (size_t)t * D;
    if (p < CAP) {
        const int e = idx[t];
        const float w = prob[t];
        const float* eot = eo + ((size_t)e * CAP + p) * D;
        f32x4 v = *(const f32x4*)(eot + j);
        f32x4 r; r[0] = w * v[0]; r[1] = w * v[1]; r[2] = w * v[2]; r[3] = w * v[3];
        *(f32x4*)(yt + j) = r;
    } else {
        *(f32x4*)(yt + j) = *(const f32x4*)(x + (size_t)t * D + j);
    }
}

// ---------------------------------------------------------------------------
// 8) aux loss scalar
// ---------------------------------------------------------------------------
__global__ void moe_loss_kernel(const float* __restrict__ Pi_sum,
                                const int* __restrict__ counts,
                                const float* __restrict__ lse2_sum,
                                float* __restrict__ out)
{
    if (threadIdx.x == 0 && blockIdx.x == 0) {
        const float invT = 1.f / (float)T_TOK;
        float dot = 0.f;
        for (int e = 0; e < E; ++e)
            dot += ((float)counts[e] * invT) * (Pi_sum[e] * invT);
        out[0] = 0.01f * (float)E * dot + 0.001f * (lse2_sum[0] * invT);
    }
}

// ---------------------------------------------------------------------------
extern "C" void kernel_launch(void* const* d_in, const int* in_sizes, int n_in,
                              void* d_out, int out_size, void* d_ws, size_t ws_size,
                              hipStream_t stream) {
    (void)in_sizes; (void)n_in; (void)out_size; (void)ws_size;
    const float* x  = (const float*)d_in[0];
    const float* Wr = (const float*)d_in[1];
    const float* W1 = (const float*)d_in[2];
    const float* b1 = (const float*)d_in[3];
    const float* W2 = (const float*)d_in[4];
    const float* b2 = (const float*)d_in[5];

    size_t off = 0;
    auto carve = [&](size_t bytes) -> char* {
        char* p = (char*)d_ws + off;
        off += (bytes + 255) & ~(size_t)255;
        return p;
    };
    _Float16* buf   = (_Float16*)carve((size_t)E * CAP * D * 2);     // 16 MB
    _Float16* h     = (_Float16*)carve((size_t)E * CAP * DFF * 2);   // 64 MB
    float* eo       = (float*)carve((size_t)E * CAP * D * 4);        // 32 MB
    _Float16* W1t   = (_Float16*)carve((size_t)E * DFF * D * 2);     // 64 MB  [E][DFF][D]
    _Float16* W2t   = (_Float16*)carve((size_t)E * D * DFF * 2);     // 64 MB  [E][D][DFF]
    float* probs8   = (float*)carve((size_t)T_TOK * E * 4);
    float* lse2t    = (float*)carve((size_t)T_TOK * 4);
    float* prob     = (float*)carve((size_t)T_TOK * 4);
    int*   idx      = (int*)carve((size_t)T_TOK * 4);
    int*   pos      = (int*)carve((size_t)T_TOK * 4);
    float* Pi_sum   = (float*)carve(E * 4);
    float* lse2_sum = (float*)carve(4);
    int*   counts   = (int*)carve(E * 4);

    float* y    = (float*)d_out;
    float* loss = (float*)d_out + (size_t)T_TOK * D;

    // 0) zero dispatch buffer
    moe_zero_kernel<<<1024, 256, 0, stream>>>((unsigned int*)buf,
                                              (size_t)E * CAP * D * 2 / 4);
    // 0b) weight convert + transpose: W1 [E][D][DFF] -> W1t [E][DFF][D] f16
    moe_wt_kernel<<<dim3(DFF / 32, D / 32, E), 256, 0, stream>>>(W1, W1t, D, DFF);
    //     W2 [E][DFF][D] -> W2t [E][D][DFF] f16
    moe_wt_kernel<<<dim3(D / 32, DFF / 32, E), 256, 0, stream>>>(W2, W2t, DFF, D);
    // 1) router
    moe_router_kernel<<<T_TOK, 256, 0, stream>>>(x, Wr, prob, idx, probs8, lse2t);
    // 2) deterministic stats
    moe_stats_kernel<<<1, 256, 0, stream>>>(probs8, lse2t, Pi_sum, lse2_sum);
    // 3) FCFS scan
    moe_scan_kernel<<<1, 32, 0, stream>>>(idx, pos, counts);
    // 4) dispatch
    moe_dispatch_kernel<<<T_TOK, 256, 0, stream>>>(x, idx, pos, buf);
    // 5) h = relu(buf @ W1 + b1)   [M=CAP, K=D, N=DFF] -> f16
    moe_gemm_kernel<true, true><<<dim3(DFF / 128, CAP / 128, E), 256, 0, stream>>>(
        buf, W1t, b1, (void*)h, CAP, D, DFF);
    // 6) eo = h @ W2 + b2          [M=CAP, K=DFF, N=D] -> f32
    moe_gemm_kernel<false, false><<<dim3(D / 128, CAP / 128, E), 256, 0, stream>>>(
        h, W2t, b2, (void*)eo, CAP, DFF, D);
    // 7) combine
    moe_combine_kernel<<<T_TOK, 256, 0, stream>>>(x, idx, pos, prob, eo, y);
    // 8) aux loss
    moe_loss_kernel<<<1, 32, 0, stream>>>(Pi_sum, counts, lse2_sum, loss);
}
